// CVRP_Decoder_84911503442123
// MI455X (gfx1250) — compile-verified, matrix operand order checked by reading, not run on hardware
//
#include <hip/hip_runtime.h>
#include <hip/hip_bf16.h>
#include <math.h>

typedef __attribute__((ext_vector_type(16))) __bf16 v16bf;
typedef __attribute__((ext_vector_type(8)))  __bf16 v8bf;
typedef __attribute__((ext_vector_type(8)))  float  v8f;
typedef __attribute__((ext_vector_type(4)))  float  v4f;

#define BB 32
#define PP 1024
#define NN 1024
#define EE 128
#define LDA 40   // LDS A-tile stride (bf16 elems): 80B rows, 16B-aligned sub-chunks
#define LDC 136  // LDS den-tile stride (f32), non-pow2 to spread banks

// Build a 16x32 (or 32x16) bf16 WMMA fragment from two contiguous 16-byte runs.
// Per ISA layout: lane half=lane>>4 selects K runs [kb+8h .. +7] and [kb+16+8h .. +7].
static __device__ __forceinline__ v16bf frag_ld(const __bf16* p0, const __bf16* p1) {
  union { v16bf v; v8bf h[2]; } u;
  u.h[0] = *(const v8bf*)p0;
  u.h[1] = *(const v8bf*)p1;
  return u.v;
}

static __device__ __forceinline__ v8f wmma_bf16(v16bf a, v16bf b, v8f c) {
  return __builtin_amdgcn_wmma_f32_16x16x32_bf16(false, a, false, b, (short)0, c, false, false);
}

// ---------------------------------------------------------------------------
// Kernel 0: transpose + convert weights to bf16: WT[h][d] = W[d][h]
// ---------------------------------------------------------------------------
__global__ __launch_bounds__(256) void wprep_kernel(
    const float* __restrict__ Wq1, const float* __restrict__ Wq2,
    const float* __restrict__ Wql, const float* __restrict__ Wk,
    const float* __restrict__ Wv,
    __bf16* __restrict__ wq1T, __bf16* __restrict__ wq2T, __bf16* __restrict__ wqlT,
    __bf16* __restrict__ wkT,  __bf16* __restrict__ wvT) {
  int idx = blockIdx.x * blockDim.x + threadIdx.x;
  if (idx >= EE * EE) return;
  int d = idx / EE, h = idx % EE;
  wq1T[h * EE + d] = (__bf16)Wq1[d * EE + h];
  wq2T[h * EE + d] = (__bf16)Wq2[d * EE + h];
  wqlT[h * EE + d] = (__bf16)Wql[d * EE + h];   // first 128 rows of [130][128]
  wkT [h * EE + d] = (__bf16)Wk [d * EE + h];
  wvT [h * EE + d] = (__bf16)Wv [d * EE + h];
}

// stage one 32x32 f32 tile -> bf16 LDS A-tile (row stride LDA)
static __device__ __forceinline__ void stage_a(__bf16* As, const float* src,
                                               int ldsrc, int tid) {
  int rr = tid >> 3;          // 0..31
  int cc = (tid & 7) * 4;     // 0..28
  v4f x = *(const v4f*)(src + (size_t)rr * ldsrc + cc);
  __bf16* d = As + rr * LDA + cc;
  d[0] = (__bf16)x[0]; d[1] = (__bf16)x[1]; d[2] = (__bf16)x[2]; d[3] = (__bf16)x[3];
}

// ---------------------------------------------------------------------------
// Kernel 1: per batch k = nodes@Wk, v = nodes@Wv via WMMA; emit
//   ekT[b][d][n]  = bf16(exp(k)),  ekvT[b][d][n] = bf16(exp(k)*v)  (transposed!)
//   nodesbf[b][n][d] = bf16(nodes)
// block: 32 node rows x 256 out cols (K||V), 8 waves, each 4 16x16 tiles
// ---------------------------------------------------------------------------
__global__ __launch_bounds__(256) void nodes_prep_kernel(
    const float* __restrict__ nodes, const __bf16* __restrict__ wkT,
    const __bf16* __restrict__ wvT,  __bf16* __restrict__ nodesbf,
    __bf16* __restrict__ ekT, __bf16* __restrict__ ekvT) {
  __shared__ __bf16 As[32 * LDA];
  __shared__ float  Cs[32 * 256];
  const int b = blockIdx.y;
  const int row0 = blockIdx.x * 32;
  const int tid = threadIdx.x, wave = tid >> 5, lane = tid & 31;
  const int half = lane >> 4, r = lane & 15;
  const int rt = wave >> 2;            // row tile 0..1
  const int ct0 = (wave & 3) * 4;      // 4 col tiles
  const float* nb = nodes + (size_t)b * NN * EE + (size_t)row0 * EE;

  v8f acc[4] = {};
  for (int kt = 0; kt < EE / 32; ++kt) {
    const int kb = kt * 32;
    stage_a(As, nb + kb, EE, tid);
    __syncthreads();
    const __bf16* arow = &As[(rt * 16 + r) * LDA];
    v16bf a = frag_ld(arow + half * 8, arow + 16 + half * 8);
#pragma unroll
    for (int i = 0; i < 4; ++i) {
      int ct = ct0 + i;
      const __bf16* wT = (ct < 8 ? wkT : wvT) + ((size_t)((ct & 7) * 16 + r)) * EE + kb;
      v16bf bf = frag_ld(wT + half * 8, wT + 16 + half * 8);
      acc[i] = wmma_bf16(a, bf, acc[i]);
    }
    __syncthreads();
  }
#pragma unroll
  for (int i = 0; i < 4; ++i) {
    int col = (ct0 + i) * 16 + r;
#pragma unroll
    for (int r2 = 0; r2 < 8; ++r2) {
      int m = rt * 16 + r2 + 8 * half;
      Cs[m * 256 + col] = acc[i][r2];
    }
  }
  __syncthreads();
  for (int e = tid; e < 32 * 128; e += 256) {
    int rr = e >> 7, d = e & 127;
    float kv = Cs[rr * 256 + d];
    float vv = Cs[rr * 256 + 128 + d];
    float ek = __expf(kv);
    int n = row0 + rr;
    ekT [(size_t)b * EE * NN + (size_t)d * NN + n] = (__bf16)ek;
    ekvT[(size_t)b * EE * NN + (size_t)d * NN + n] = (__bf16)(ek * vv);
    nodesbf[(size_t)b * NN * EE + (size_t)n * EE + d] = (__bf16)nb[rr * EE + d];
  }
}

// ---------------------------------------------------------------------------
// Kernel 2: q = q1@Wq1 + q2@Wq2 + last@Wql[0:128] (+ rank-2 tail); sigq = sigmoid(q)
// block: 32 rows x 128 cols, 8 waves x 2 tiles
// ---------------------------------------------------------------------------
__global__ __launch_bounds__(256) void q_prep_kernel(
    const float* __restrict__ q1, const float* __restrict__ q2,
    const float* __restrict__ lastn, const float* __restrict__ loadv,
    const float* __restrict__ leftv, const float* __restrict__ Wql,
    const __bf16* __restrict__ wq1T, const __bf16* __restrict__ wq2T,
    const __bf16* __restrict__ wqlT, float* __restrict__ sigq) {
  __shared__ __bf16 As[32 * LDA];
  const int b = blockIdx.y;
  const int row0 = blockIdx.x * 32;
  const int tid = threadIdx.x, wave = tid >> 5, lane = tid & 31;
  const int half = lane >> 4, r = lane & 15;
  const int rt = wave >> 2;
  const int ct0 = (wave & 3) * 2;      // 2 col tiles per wave
  const size_t off = (size_t)b * PP * EE + (size_t)row0 * EE;
  const float* srcs[3] = { q1 + off, q2 + off, lastn + off };
  const __bf16* wts[3] = { wq1T, wq2T, wqlT };

  v8f acc[2] = {};
  for (int s = 0; s < 3; ++s) {
    for (int kt = 0; kt < EE / 32; ++kt) {
      const int kb = kt * 32;
      stage_a(As, srcs[s] + kb, EE, tid);
      __syncthreads();
      const __bf16* arow = &As[(rt * 16 + r) * LDA];
      v16bf a = frag_ld(arow + half * 8, arow + 16 + half * 8);
#pragma unroll
      for (int i = 0; i < 2; ++i) {
        const __bf16* wT = wts[s] + ((size_t)((ct0 + i) * 16 + r)) * EE + kb;
        v16bf bf = frag_ld(wT + half * 8, wT + 16 + half * 8);
        acc[i] = wmma_bf16(a, bf, acc[i]);
      }
      __syncthreads();
    }
  }
#pragma unroll
  for (int i = 0; i < 2; ++i) {
    int h = (ct0 + i) * 16 + r;
    float w128 = Wql[128 * EE + h];
    float w129 = Wql[129 * EE + h];
#pragma unroll
    for (int r2 = 0; r2 < 8; ++r2) {
      int p = row0 + rt * 16 + r2 + 8 * half;
      float qv = acc[i][r2] + loadv[b * PP + p] * w128 + leftv[b * PP + p] * w129;
      sigq[(size_t)b * PP * EE + (size_t)p * EE + h] = 1.f / (1.f + __expf(-qv));
    }
  }
}

// ---------------------------------------------------------------------------
// Kernel 3: AFT core.  e_bias = exp(ninf - c1*cur_dist) streamed into
// double-buffered bf16 LDS A-tiles; num||den = e_bias @ [ekvT || ekT]^T.
// Wave mapping: each wave covers BOTH row tiles x a unique col-tile pair
// (ct0 = wave*2) -> every B fragment loaded exactly once per block.
// Epilogue: waves 4-7 (den) exchange via LDS; waves 0-3 fuse sig*num/den.
// ---------------------------------------------------------------------------
__global__ __launch_bounds__(256) void aft_gemm_kernel(
    const float* __restrict__ cdist, const float* __restrict__ ninf,
    const float* __restrict__ logsc, const float* __restrict__ aalp,
    const __bf16* __restrict__ ekvT, const __bf16* __restrict__ ekT,
    const float* __restrict__ sigq, __bf16* __restrict__ aftbf) {
  __shared__ __bf16 As[2][32 * LDA];
  __shared__ float  Cden[32 * LDC];
  const int b = blockIdx.y;
  const int row0 = blockIdx.x * 32;
  const int tid = threadIdx.x, wave = tid >> 5, lane = tid & 31;
  const int half = lane >> 4, r = lane & 15;
  const int ct0 = wave * 2;            // 2 unique col tiles; both row tiles
  const float c1 = logsc[0] * aalp[0];
  const float* cdp = cdist + (size_t)b * PP * NN + (size_t)row0 * NN;
  const float* nfp = ninf  + (size_t)b * PP * NN + (size_t)row0 * NN;
  const __bf16* ekvb = ekvT + (size_t)b * EE * NN;
  const __bf16* ekb  = ekT  + (size_t)b * EE * NN;

  const int rr = tid >> 3, cc = (tid & 7) * 4;
  const float* cdrow = cdp + (size_t)rr * NN + cc;
  const float* nfrow = nfp + (size_t)rr * NN + cc;

  // prologue: stage k-tile 0 into buffer 0
  {
    v4f cd4 = *(const v4f*)cdrow;
    v4f nf4 = *(const v4f*)nfrow;
    __bf16* d = &As[0][rr * LDA + cc];
    d[0] = (__bf16)__expf(nf4[0] - c1 * cd4[0]);
    d[1] = (__bf16)__expf(nf4[1] - c1 * cd4[1]);
    d[2] = (__bf16)__expf(nf4[2] - c1 * cd4[2]);
    d[3] = (__bf16)__expf(nf4[3] - c1 * cd4[3]);
  }

  v8f acc[2][2] = {};                  // [rt][i]
  for (int kt = 0; kt < NN / 32; ++kt) {
    __syncthreads();                   // staging of As[kt&1] complete (WAR safe)
    const bool more = (kt + 1) < NN / 32;
    v4f cd4 = {}, nf4 = {};
    if (more) {                        // issue next tile's global loads early
      cd4 = *(const v4f*)(cdrow + (kt + 1) * 32);
      nf4 = *(const v4f*)(nfrow + (kt + 1) * 32);
    }
    const __bf16* abuf = As[kt & 1];
    const int kb = kt * 32;
    v16bf afr[2];
#pragma unroll
    for (int rt = 0; rt < 2; ++rt) {
      const __bf16* arow = abuf + (rt * 16 + r) * LDA;
      afr[rt] = frag_ld(arow + half * 8, arow + 16 + half * 8);
    }
#pragma unroll
    for (int i = 0; i < 2; ++i) {
      int ct = ct0 + i;
      const __bf16* bp = (ct < 8 ? ekvb : ekb) + ((size_t)((ct & 7) * 16 + r)) * NN + kb;
      v16bf bfr = frag_ld(bp + half * 8, bp + 16 + half * 8);
#pragma unroll
      for (int rt = 0; rt < 2; ++rt)
        acc[rt][i] = wmma_bf16(afr[rt], bfr, acc[rt][i]);
    }
    if (more) {                        // exp + store into the other buffer
      __bf16* d = &As[(kt + 1) & 1][rr * LDA + cc];
      d[0] = (__bf16)__expf(nf4[0] - c1 * cd4[0]);
      d[1] = (__bf16)__expf(nf4[1] - c1 * cd4[1]);
      d[2] = (__bf16)__expf(nf4[2] - c1 * cd4[2]);
      d[3] = (__bf16)__expf(nf4[3] - c1 * cd4[3]);
    }
  }

  // epilogue: waves 4-7 hold den (cols 128..255) -> LDS
  if (wave >= 4) {
#pragma unroll
    for (int i = 0; i < 2; ++i) {
      int dcol = (ct0 + i - 8) * 16 + r;
#pragma unroll
      for (int rt = 0; rt < 2; ++rt)
#pragma unroll
        for (int r2 = 0; r2 < 8; ++r2) {
          int m = rt * 16 + r2 + 8 * half;
          Cden[m * LDC + dcol] = acc[rt][i][r2];
        }
    }
  }
  __syncthreads();
  // waves 0-3 hold num: AFT_out = sigq * num / (den + 1e-20) -> bf16
  if (wave < 4) {
#pragma unroll
    for (int i = 0; i < 2; ++i) {
      int dcol = (ct0 + i) * 16 + r;
#pragma unroll
      for (int rt = 0; rt < 2; ++rt)
#pragma unroll
        for (int r2 = 0; r2 < 8; ++r2) {
          int m = rt * 16 + r2 + 8 * half;
          int p = row0 + m;
          float num = acc[rt][i][r2];
          float den = Cden[m * LDC + dcol] + 1e-20f;
          float sg = sigq[(size_t)b * PP * EE + (size_t)p * EE + dcol];
          aftbf[(size_t)b * PP * EE + (size_t)p * EE + dcol] = (__bf16)(sg * num / den);
        }
    }
  }
}

// ---------------------------------------------------------------------------
// Kernel 4: score = AFT_out @ nodes^T, tanh clip, fused row softmax -> probs
// block: 16 rows x 1024 cols, 8 waves x 8 tiles; reductions shfl_xor(<=8)+LDS
// ---------------------------------------------------------------------------
__global__ __launch_bounds__(256) void score_softmax_kernel(
    const __bf16* __restrict__ aftbf, const __bf16* __restrict__ nodesbf,
    const float* __restrict__ cdist, const float* __restrict__ ninf,
    const float* __restrict__ logsc, const float* __restrict__ palp,
    float* __restrict__ out) {
  __shared__ float redA[8 * 16];
  __shared__ float redB[8 * 16];
  const int b = blockIdx.y;
  const int prow0 = blockIdx.x * 16;
  const int tid = threadIdx.x, wave = tid >> 5, lane = tid & 31;
  const int half = lane >> 4, r = lane & 15;
  const int ct0 = wave * 8;                      // 8 col tiles -> cols wave*128..+127
  const __bf16* ab = aftbf + (size_t)b * PP * EE + (size_t)(prow0 + r) * EE;
  const __bf16* nbb = nodesbf + (size_t)b * NN * EE;
  const float inv_sqrt_e = 1.0f / 11.313708498984761f;
  const float c2 = logsc[0] * palp[0];

  v8f acc[8] = {};
  for (int kt = 0; kt < EE / 32; ++kt) {
    const int kb = kt * 32;
    v16bf a = frag_ld(ab + kb + half * 8, ab + kb + 16 + half * 8);
#pragma unroll
    for (int i = 0; i < 8; ++i) {
      const __bf16* np = nbb + ((size_t)((ct0 + i) * 16 + r)) * EE + kb;
      v16bf bf = frag_ld(np + half * 8, np + 16 + half * 8);
      acc[i] = wmma_bf16(a, bf, acc[i]);
    }
  }
  // clip + mask (overwrite accumulators with final logits)
#pragma unroll
  for (int i = 0; i < 8; ++i) {
    int n = (ct0 + i) * 16 + r;
#pragma unroll
    for (int r2 = 0; r2 < 8; ++r2) {
      int p = prow0 + r2 + 8 * half;
      size_t idx = (size_t)b * PP * NN + (size_t)p * NN + n;
      float sv = acc[i][r2] * inv_sqrt_e - c2 * cdist[idx];
      acc[i][r2] = 10.0f * tanhf(sv) + ninf[idx];
    }
  }
  // row max (rows split per half: masks 1..8 stay inside the 16-lane half)
  float part[8], rowmax[8], rowsum[8];
#pragma unroll
  for (int r2 = 0; r2 < 8; ++r2) {
    float m = acc[0][r2];
#pragma unroll
    for (int i = 1; i < 8; ++i) m = fmaxf(m, acc[i][r2]);
    for (int s = 1; s < 16; s <<= 1) m = fmaxf(m, __shfl_xor(m, s, 32));
    part[r2] = m;
  }
  if (r == 0)
#pragma unroll
    for (int r2 = 0; r2 < 8; ++r2) redA[wave * 16 + half * 8 + r2] = part[r2];
  __syncthreads();
#pragma unroll
  for (int r2 = 0; r2 < 8; ++r2) {
    float m = redA[0 * 16 + half * 8 + r2];
    for (int w = 1; w < 8; ++w) m = fmaxf(m, redA[w * 16 + half * 8 + r2]);
    rowmax[r2] = m;
  }
  // row sum of exp
#pragma unroll
  for (int r2 = 0; r2 < 8; ++r2) {
    float s = 0.f;
#pragma unroll
    for (int i = 0; i < 8; ++i) s += __expf(acc[i][r2] - rowmax[r2]);
    for (int t = 1; t < 16; t <<= 1) s += __shfl_xor(s, t, 32);
    part[r2] = s;
  }
  if (r == 0)
#pragma unroll
    for (int r2 = 0; r2 < 8; ++r2) redB[wave * 16 + half * 8 + r2] = part[r2];
  __syncthreads();
#pragma unroll
  for (int r2 = 0; r2 < 8; ++r2) {
    float s = 0.f;
    for (int w = 0; w < 8; ++w) s += redB[w * 16 + half * 8 + r2];
    rowsum[r2] = s;
  }
  // normalized probs
#pragma unroll
  for (int i = 0; i < 8; ++i) {
    int n = (ct0 + i) * 16 + r;
#pragma unroll
    for (int r2 = 0; r2 < 8; ++r2) {
      int p = prow0 + r2 + 8 * half;
      out[(size_t)b * PP * NN + (size_t)p * NN + n] =
          __expf(acc[i][r2] - rowmax[r2]) / rowsum[r2];
    }
  }
}

// ---------------------------------------------------------------------------
extern "C" void kernel_launch(void* const* d_in, const int* in_sizes, int n_in,
                              void* d_out, int out_size, void* d_ws, size_t ws_size,
                              hipStream_t stream) {
  const float* nodes = (const float*)d_in[0];
  const float* q1    = (const float*)d_in[1];
  const float* q2    = (const float*)d_in[2];
  const float* lastn = (const float*)d_in[3];
  const float* loadv = (const float*)d_in[4];
  const float* leftv = (const float*)d_in[5];
  const float* cdist = (const float*)d_in[6];
  const float* logsc = (const float*)d_in[7];
  const float* ninf  = (const float*)d_in[8];
  const float* Wq1   = (const float*)d_in[9];
  const float* Wq2   = (const float*)d_in[10];
  const float* Wql   = (const float*)d_in[11];
  const float* Wk    = (const float*)d_in[12];
  const float* Wv    = (const float*)d_in[13];
  const float* aalp  = (const float*)d_in[14];
  const float* palp  = (const float*)d_in[15];
  float* out = (float*)d_out;
  (void)in_sizes; (void)n_in; (void)out_size;

  char* ws = (char*)d_ws;
  size_t o = 0;
  auto take = [&](size_t bytes) -> char* {
    char* p = ws + o;
    o = (o + bytes + 255) & ~(size_t)255;
    return p;
  };
  __bf16* ekvT    = (__bf16*)take((size_t)BB * EE * NN * 2);  // 8 MB
  __bf16* ekT     = (__bf16*)take((size_t)BB * EE * NN * 2);  // 8 MB
  __bf16* nodesbf = (__bf16*)take((size_t)BB * NN * EE * 2);  // 8 MB
  __bf16* aftbf   = (__bf16*)take((size_t)BB * PP * EE * 2);  // 8 MB
  float*  sigq    = (float*) take((size_t)BB * PP * EE * 4);  // 16 MB
  __bf16* wq1T = (__bf16*)take((size_t)EE * EE * 2);
  __bf16* wq2T = (__bf16*)take((size_t)EE * EE * 2);
  __bf16* wqlT = (__bf16*)take((size_t)EE * EE * 2);
  __bf16* wkT  = (__bf16*)take((size_t)EE * EE * 2);
  __bf16* wvT  = (__bf16*)take((size_t)EE * EE * 2);
  (void)ws_size; // ~51 MB used

  wprep_kernel<<<(EE * EE + 255) / 256, 256, 0, stream>>>(
      Wq1, Wq2, Wql, Wk, Wv, wq1T, wq2T, wqlT, wkT, wvT);
  nodes_prep_kernel<<<dim3(NN / 32, BB), 256, 0, stream>>>(
      nodes, wkT, wvT, nodesbf, ekT, ekvT);
  q_prep_kernel<<<dim3(PP / 32, BB), 256, 0, stream>>>(
      q1, q2, lastn, loadv, leftv, Wql, wq1T, wq2T, wqlT, sigq);
  aft_gemm_kernel<<<dim3(PP / 32, BB), 256, 0, stream>>>(
      cdist, ninf, logsc, aalp, ekvT, ekT, sigq, aftbf);
  score_softmax_kernel<<<dim3(PP / 16, BB), 256, 0, stream>>>(
      aftbf, nodesbf, cdist, ninf, logsc, palp, out);
}
